// Model_88055419502656
// MI455X (gfx1250) — matrix operand order repaired and checked by reading, not compiled, not run on hardware
//
#include <hip/hip_runtime.h>
#include <cstdint>

typedef __attribute__((ext_vector_type(16))) __bf16 v16bf;
typedef __attribute__((ext_vector_type(8)))  float  v8f;
typedef unsigned int v4u __attribute__((ext_vector_type(4)));
typedef int v8i_ __attribute__((ext_vector_type(8)));
typedef int v4i_ __attribute__((ext_vector_type(4)));

#define DMODEL 512
#define NHEADS 8
#define DH     64
#define DFF    2048
#define LTOT   2048
#define NBATCH 8
#define NBH    64
#define NHASH  4
#define NBUCK  512
#define NCHUNK 2048            // NHASH * NBUCK
#define SORTLEN 8192           // NHASH * LTOT
#define MROWS  (NBATCH*LTOT)   // 16384

// ---------- helpers ----------
__device__ __forceinline__ unsigned short f2bfu(float f) {
  union { float f; unsigned u; } a; a.f = f;
  unsigned u = a.u;
  return (unsigned short)((u + 0x7FFFu + ((u >> 16) & 1u)) >> 16);
}

union AFrag { v16bf v; unsigned u[8]; };

// ---------- TDM: 2D bf16 tile (tile_d1 rows x tile_d0 cols) global -> LDS ----------
// D# packing per CDNA5 ISA 8.3/8.4: group0 {count=1, lds_addr, global_addr, type=2},
// group1 {data_size=2B, tensor dims, tile dims, dim0 stride}.
__device__ __forceinline__ void tdm_load_tile(const unsigned short* gbase, unsigned ldsoff,
                                              unsigned tensor_d0 /*row len (elems)*/,
                                              unsigned tensor_d1 /*rows*/,
                                              unsigned tile_d0, unsigned tile_d1) {
  unsigned long long ga = (unsigned long long)(uintptr_t)gbase;
  v4u g0;
  g0[0] = 1u;                                               // count=1, user mode
  g0[1] = ldsoff;                                           // lds_addr
  g0[2] = (unsigned)(ga & 0xffffffffu);                     // global_addr[31:0]
  g0[3] = (unsigned)((ga >> 32) & 0x01ffffffu) | (2u << 30);// global_addr[56:32] | type=2
  v8i_ g1;
  g1[0] = (int)(1u << 16);                                  // workgroup_mask=0, data_size=1 (2 bytes)
  g1[1] = (int)((tensor_d0 & 0xffffu) << 16);               // tensor_dim0[15:0] at bits[63:48]
  g1[2] = (int)((tensor_d0 >> 16) | ((tensor_d1 & 0xffffu) << 16)); // td0 hi | td1 lo
  g1[3] = (int)((tensor_d1 >> 16) | (tile_d0 << 16));       // td1 hi | tile_dim0
  g1[4] = (int)(tile_d1 & 0xffffu);                         // tile_dim1 (tile_dim2=0)
  g1[5] = (int)tensor_d0;                                   // tensor_dim0_stride[31:0]
  g1[6] = 0;                                                // stride hi16 | dim1_stride lo16
  g1[7] = 0;
  v4i_ z4 = {0, 0, 0, 0};
#if __clang_major__ >= 23
  v8i_ z8 = {0, 0, 0, 0, 0, 0, 0, 0};
  __builtin_amdgcn_tensor_load_to_lds(g0, g1, z4, z4, z8, 0);
#else
  __builtin_amdgcn_tensor_load_to_lds(g0, g1, z4, z4, 0);
#endif
}

// ---------- weight fp32 -> bf16 ----------
__global__ void cvt_bf16_kernel(const float* __restrict__ s, unsigned short* __restrict__ d, int n) {
  int i = blockIdx.x * blockDim.x + threadIdx.x;
  if (i < n) d[i] = f2bfu(s[i]);
}

// ---------- embedding: token conv1d(circular) + temporal linear + sinusoidal PE ----------
__global__ __launch_bounds__(128) void embed_kernel(
    const float* __restrict__ x_enc, const float* __restrict__ x_dec,
    const float* __restrict__ xm_enc, const float* __restrict__ xm_dec,
    const float* __restrict__ token_w, const float* __restrict__ temp_w,
    float* __restrict__ h, unsigned short* __restrict__ hbf) {
  int bl = blockIdx.x;
  int b = bl >> 11, l = bl & (LTOT - 1);
  __shared__ float sx[3][7];
  __shared__ float sxm[4];
  int t = threadIdx.x;
  if (t < 21) {
    int k = t / 7, c = t % 7;
    int li = (l + k - 1 + LTOT) & (LTOT - 1);
    float v;
    if (li < 1536) v = x_enc[(b * 1536 + li) * 7 + c];
    else           v = x_dec[(b * 768 + (li - 1536 + 256)) * 7 + c];
    sx[k][c] = v;
  } else if (t >= 32 && t < 36) {
    int c = t - 32;
    float v;
    if (l < 1536) v = xm_enc[(b * 1536 + l) * 4 + c];
    else          v = xm_dec[(b * 768 + (l - 1536 + 256)) * 4 + c];
    sxm[c] = v;
  }
  __syncthreads();
  for (int d = t; d < DMODEL; d += 128) {
    float acc = 0.f;
    #pragma unroll
    for (int c = 0; c < 7; ++c)
      #pragma unroll
      for (int k = 0; k < 3; ++k)
        acc += sx[k][c] * token_w[(d * 7 + c) * 3 + k];
    #pragma unroll
    for (int c = 0; c < 4; ++c) acc += sxm[c] * temp_w[d * 4 + c];
    int i = d >> 1;
    float div = expf(-(float)(2 * i) * (9.210340371976184f / 512.0f));
    float ang = (float)l * div;
    acc += (d & 1) ? cosf(ang) : sinf(ang);
    h[(size_t)bl * DMODEL + d] = acc;
    hbf[(size_t)bl * DMODEL + d] = f2bfu(acc);
  }
}

// ---------- bf16 WMMA GEMM with TDM double-buffered tiles ----------
// C[M,N] = A[M,K](bf16) * W[N,K]^T ; flags: 1=bias, 2=residual(fp32), 4=gelu, 8=write bf16 to Cbf
__global__ __launch_bounds__(256) void gemm_bf16_tdm(
    const unsigned short* __restrict__ A, const unsigned short* __restrict__ W,
    float* C, unsigned short* Cbf, const float* __restrict__ bias, const float* res,
    int M, int N, int K, int flags) {
  __shared__ unsigned short As[2][128 * 64];
  __shared__ unsigned short Bs[2][64 * 64];
  int t = threadIdx.x;
  int lane = t & 31, wid = t >> 5;
  int m0blk = blockIdx.y * 128;
  int n0blk = blockIdx.x * 64;
  int wm = (wid >> 1) * 32;       // wave M offset in tile
  int wn = (wid & 1) * 32;        // wave N offset in tile
  int hv = lane >> 4;             // lane half
  int l16 = lane & 15;

  v8f acc[2][2];
  #pragma unroll
  for (int i = 0; i < 2; ++i)
    #pragma unroll
    for (int j = 0; j < 2; ++j)
      #pragma unroll
      for (int r = 0; r < 8; ++r) acc[i][j][r] = 0.0f;

  unsigned ldsA[2] = { (unsigned)(uintptr_t)&As[0][0], (unsigned)(uintptr_t)&As[1][0] };
  unsigned ldsB[2] = { (unsigned)(uintptr_t)&Bs[0][0], (unsigned)(uintptr_t)&Bs[1][0] };
  int nstage = K >> 6;

  if (wid == 0) {
    tdm_load_tile(A + (size_t)m0blk * K, ldsA[0], (unsigned)K, 128u, 64u, 128u);
    tdm_load_tile(W + (size_t)n0blk * K, ldsB[0], (unsigned)K, 64u, 64u, 64u);
  }

  for (int s = 0; s < nstage; ++s) {
    if (wid == 0) __builtin_amdgcn_s_wait_tensorcnt(0);
    __syncthreads();                          // stage-s tiles landed; prev compute done
    if (wid == 0 && s + 1 < nstage) {
      int nb = (s + 1) & 1;
      tdm_load_tile(A + (size_t)m0blk * K + (s + 1) * 64, ldsA[nb], (unsigned)K, 128u, 64u, 128u);
      tdm_load_tile(W + (size_t)n0blk * K + (s + 1) * 64, ldsB[nb], (unsigned)K, 64u, 64u, 64u);
    }
    int bsel = s & 1;
    #pragma unroll
    for (int ks = 0; ks < 2; ++ks) {
      AFrag af[2], bf_[2];
      #pragma unroll
      for (int sM = 0; sM < 2; ++sM) {
        const unsigned* uA = (const unsigned*)(&As[bsel][(wm + sM * 16 + l16) * 64 + ks * 32]);
        #pragma unroll
        for (int i = 0; i < 4; ++i) {
          af[sM].u[i]     = uA[hv * 4 + i];        // K: half*8 + 0..7
          af[sM].u[4 + i] = uA[8 + hv * 4 + i];    // K: 16 + half*8 + 0..7
        }
      }
      #pragma unroll
      for (int sN = 0; sN < 2; ++sN) {
        const unsigned* uB = (const unsigned*)(&Bs[bsel][(wn + sN * 16 + l16) * 64 + ks * 32]);
        #pragma unroll
        for (int i = 0; i < 4; ++i) {
          bf_[sN].u[i]     = uB[hv * 4 + i];        // K: half*8 + 0..7
          bf_[sN].u[4 + i] = uB[8 + hv * 4 + i];    // K: 16 + half*8 + 0..7
        }
      }
      #pragma unroll
      for (int sM = 0; sM < 2; ++sM)
        #pragma unroll
        for (int sN = 0; sN < 2; ++sN)
          acc[sM][sN] = __builtin_amdgcn_wmma_f32_16x16x32_bf16(
              false, af[sM].v, false, bf_[sN].v, (short)0, acc[sM][sN], false, false);
    }
    __syncthreads();                          // all reads of buf[bsel] done before TDM reuses it
  }

  #pragma unroll
  for (int sM = 0; sM < 2; ++sM)
    #pragma unroll
    for (int sN = 0; sN < 2; ++sN)
      #pragma unroll
      for (int r = 0; r < 8; ++r) {
        int row = m0blk + wm + sM * 16 + hv * 8 + r;
        int col = n0blk + wn + sN * 16 + l16;
        float v = acc[sM][sN][r];
        if (flags & 1) v += bias[col];
        if (flags & 4) v = 0.5f * v * (1.0f + erff(v * 0.7071067811865475f));
        if (flags & 2) v += res[(size_t)row * N + col];
        if (flags & 8) Cbf[(size_t)row * N + col] = f2bfu(v);
        else           C[(size_t)row * N + col] = v;
      }
}

// ---------- LSH hashing: argmax over [R, -R] rotations, per (bh, l, hash) ----------
__global__ __launch_bounds__(128) void hash_kernel(
    const float* __restrict__ qk, const float* __restrict__ rot, int* __restrict__ buckets) {
  int bl = blockIdx.x;
  int bh = bl >> 11, l = bl & (LTOT - 1);
  __shared__ float q[64];
  int t = threadIdx.x;
  if (t < 64)
    q[t] = qk[((size_t)((bh >> 3) * LTOT + l)) * DMODEL + (bh & 7) * DH + t];
  __syncthreads();
  int h = t >> 5, lane = t & 31;   // wave h handles hash round h
  float best = -1e30f; int bidx = 0x7fffffff;
  for (int j = 0; j < 8; ++j) {
    int r = lane + j * 32;
    float d = 0.f;
    for (int kk = 0; kk < 64; ++kk) d += q[kk] * rot[(size_t)(kk * NHASH + h) * 256 + r];
    if (d > best || (d == best && r < bidx)) { best = d; bidx = r; }
    float dn = -d; int rn = r + 256;
    if (dn > best || (dn == best && rn < bidx)) { best = dn; bidx = rn; }
  }
  for (int o = 16; o >= 1; o >>= 1) {
    float ob = __shfl_xor(best, o, 32);
    int   oi = __shfl_xor(bidx, o, 32);
    if (ob > best || (ob == best && oi < bidx)) { best = ob; bidx = oi; }
  }
  if (lane == 0) buckets[(size_t)bh * SORTLEN + h * LTOT + l] = bidx + h * NBUCK;
}

// ---------- counting sort phase 1: histogram + exclusive scan per BH ----------
__global__ __launch_bounds__(256) void hist_kernel(
    const int* __restrict__ buckets, int* __restrict__ start) {
  int bh = blockIdx.x;
  __shared__ int hist[2048];
  __shared__ int part[256];
  int t = threadIdx.x;
  for (int i = t; i < 2048; i += 256) hist[i] = 0;
  __syncthreads();
  for (int i = t; i < SORTLEN; i += 256)
    atomicAdd(&hist[buckets[(size_t)bh * SORTLEN + i]], 1);
  __syncthreads();
  int s = 0;
  #pragma unroll
  for (int k = 0; k < 8; ++k) s += hist[t * 8 + k];
  part[t] = s;
  __syncthreads();
  if (t == 0) {
    int run = 0;
    for (int i = 0; i < 256; ++i) { int tmp = part[i]; part[i] = run; run += tmp; }
  }
  __syncthreads();
  int run = part[t];
  #pragma unroll
  for (int k = 0; k < 8; ++k) {
    int b = t * 8 + k;
    int c = hist[b];
    start[(size_t)bh * 2048 + b] = run;
    run += c;
  }
}

// ---------- counting sort phase 2: stable scatter ----------
__global__ void scatter_kernel(const int* __restrict__ buckets, const int* __restrict__ start,
                               int* __restrict__ st, int* __restrict__ undo) {
  int gid = blockIdx.x * blockDim.x + threadIdx.x;
  if (gid >= NBH * 2048) return;
  int bh = gid >> 11, b = gid & 2047;
  int h = b >> 9;
  const int* Bk = buckets + (size_t)bh * SORTLEN + h * LTOT;
  int p = start[(size_t)bh * 2048 + b];
  for (int t2 = 0; t2 < LTOT; ++t2) {
    if (Bk[t2] == b) {
      st[(size_t)bh * SORTLEN + p] = t2;
      undo[(size_t)bh * SORTLEN + h * LTOT + t2] = p;
      ++p;
    }
  }
}

// ---------- chunked LSH attention: one wave per (bh, chunk) ----------
__global__ __launch_bounds__(256) void chunk_attn_kernel(
    const float* __restrict__ qk, const float* __restrict__ v,
    const int* __restrict__ st, float* __restrict__ so, float* __restrict__ slog) {
  __shared__ float SQ[8][256];
  __shared__ float SK[8][512];
  __shared__ float SV[8][512];
  __shared__ float SP[8][32];
  int t = threadIdx.x, wid = t >> 5, lane = t & 31;
  int gci = blockIdx.x * 8 + wid;
  int bh = gci >> 11, c = gci & (NCHUNK - 1);
  size_t rowbase = (size_t)(bh >> 3) * LTOT;
  int hoff = (bh & 7) * DH;

  int tq[4], tk[8];
  #pragma unroll
  for (int i = 0; i < 4; ++i) tq[i] = st[(size_t)bh * SORTLEN + c * 4 + i];
  #pragma unroll
  for (int j = 0; j < 8; ++j) {
    int cc = (c + ((j < 4) ? 0 : (NCHUNK - 1))) & (NCHUNK - 1);
    tk[j] = st[(size_t)bh * SORTLEN + cc * 4 + (j & 3)];
  }
  float* Q = SQ[wid]; float* Kl = SK[wid]; float* Vl = SV[wid]; float* P = SP[wid];
  #pragma unroll
  for (int m = 0; m < 8; ++m) {
    int e = lane + m * 32; int i = e >> 6; int dd = e & 63;
    Q[e] = qk[(rowbase + tq[i]) * DMODEL + hoff + dd];
  }
  #pragma unroll
  for (int m = 0; m < 16; ++m) {
    int e = lane + m * 32; int j = e >> 6; int dd = e & 63;
    Kl[e] = qk[(rowbase + tk[j]) * DMODEL + hoff + dd];
    Vl[e] = v [(rowbase + tk[j]) * DMODEL + hoff + dd];
  }
  #pragma unroll
  for (int j = 0; j < 8; ++j) {
    float a0 = Kl[j * 64 + lane], a1 = Kl[j * 64 + lane + 32];
    float ss = a0 * a0 + a1 * a1;
    for (int o = 16; o >= 1; o >>= 1) ss += __shfl_xor(ss, o, 32);
    float inv = 1.0f / fmaxf(sqrtf(ss), 1e-12f);
    Kl[j * 64 + lane] = a0 * inv;
    Kl[j * 64 + lane + 32] = a1 * inv;
  }
  int qi = lane >> 3, kj = lane & 7;
  float d = 0.f;
  for (int kk = 0; kk < 64; ++kk) d += Q[qi * 64 + kk] * Kl[kj * 64 + kk];
  d *= 0.125f;
  if (tq[qi] == tk[kj]) d = -5.0e4f;
  float mx = d;
  #pragma unroll
  for (int o = 1; o < 8; o <<= 1) mx = fmaxf(mx, __shfl_xor(mx, o, 32));
  float ex = expf(d - mx);
  float sm = ex;
  #pragma unroll
  for (int o = 1; o < 8; o <<= 1) sm += __shfl_xor(sm, o, 32);
  float lse = mx + logf(sm);
  P[lane] = expf(d - lse);
  int i2 = lane >> 3, dbase = (lane & 7) * 8;
  float o8[8];
  #pragma unroll
  for (int dd = 0; dd < 8; ++dd) o8[dd] = 0.f;
  #pragma unroll
  for (int j = 0; j < 8; ++j) {
    float pj = P[i2 * 8 + j];
    #pragma unroll
    for (int dd = 0; dd < 8; ++dd) o8[dd] += pj * Vl[j * 64 + dbase + dd];
  }
  float* outp = so + ((size_t)bh * SORTLEN + c * 4 + i2) * DH + dbase;
  #pragma unroll
  for (int dd = 0; dd < 8; ++dd) outp[dd] = o8[dd];
  if ((lane & 7) == 0) slog[(size_t)bh * SORTLEN + c * 4 + i2] = lse;
}

// ---------- unsort + softmax-combine hash rounds, merge heads (bf16 out) ----------
__global__ void combine_kernel(const float* __restrict__ so, const float* __restrict__ slog,
                               const int* __restrict__ undo, unsigned short* __restrict__ attnbf) {
  int gid = blockIdx.x * blockDim.x + threadIdx.x;
  if (gid >= NBH * LTOT) return;
  int bh = gid >> 11, l = gid & (LTOT - 1);
  int u[4]; float lg[4];
  #pragma unroll
  for (int h = 0; h < 4; ++h) {
    u[h] = undo[(size_t)bh * SORTLEN + h * LTOT + l];
    lg[h] = slog[(size_t)bh * SORTLEN + u[h]];
  }
  float mx = fmaxf(fmaxf(lg[0], lg[1]), fmaxf(lg[2], lg[3]));
  float w[4], s = 0.f;
  #pragma unroll
  for (int h = 0; h < 4; ++h) { w[h] = expf(lg[h] - mx); s += w[h]; }
  float inv = 1.0f / s;
  #pragma unroll
  for (int h = 0; h < 4; ++h) w[h] *= inv;
  unsigned short* dst = attnbf + ((size_t)(bh >> 3) * LTOT + l) * DMODEL + (bh & 7) * DH;
  for (int dd = 0; dd < DH; dd += 4) {
    float4 acc = make_float4(0.f, 0.f, 0.f, 0.f);
    #pragma unroll
    for (int h = 0; h < 4; ++h) {
      const float4 sv = *(const float4*)(so + ((size_t)bh * SORTLEN + u[h]) * DH + dd);
      acc.x += w[h] * sv.x; acc.y += w[h] * sv.y;
      acc.z += w[h] * sv.z; acc.w += w[h] * sv.w;
    }
    dst[dd + 0] = f2bfu(acc.x); dst[dd + 1] = f2bfu(acc.y);
    dst[dd + 2] = f2bfu(acc.z); dst[dd + 3] = f2bfu(acc.w);
  }
}

// ---------- layernorm: one wave per row of 512; optional bf16 copy ----------
__global__ __launch_bounds__(256) void ln_kernel(const float* src, float* dst,
                                                 unsigned short* dstbf,
                                                 const float* __restrict__ g,
                                                 const float* __restrict__ b, int nrows) {
  int wid = (blockIdx.x * blockDim.x + threadIdx.x) >> 5;
  int lane = threadIdx.x & 31;
  if (wid >= nrows) return;
  const float* x = src + (size_t)wid * DMODEL;
  float vals[16];
  float s = 0.f;
  #pragma unroll
  for (int j = 0; j < 16; ++j) { vals[j] = x[lane + j * 32]; s += vals[j]; }
  for (int o = 16; o >= 1; o >>= 1) s += __shfl_xor(s, o, 32);
  float mu = s * (1.0f / 512.0f);
  float vs = 0.f;
  #pragma unroll
  for (int j = 0; j < 16; ++j) { float dv = vals[j] - mu; vs += dv * dv; }
  for (int o = 16; o >= 1; o >>= 1) vs += __shfl_xor(vs, o, 32);
  float inv = rsqrtf(vs * (1.0f / 512.0f) + 1e-5f);
  float* y = dst + (size_t)wid * DMODEL;
  #pragma unroll
  for (int j = 0; j < 16; ++j) {
    int col = lane + j * 32;
    float r = (vals[j] - mu) * inv * g[col] + b[col];
    y[col] = r;
    if (dstbf) dstbf[(size_t)wid * DMODEL + col] = f2bfu(r);
  }
}

// ---------- final projection: last PRED_LEN rows ----------
__global__ __launch_bounds__(256) void proj_kernel(const float* __restrict__ h,
                                                   const float* __restrict__ pw,
                                                   const float* __restrict__ pb,
                                                   float* __restrict__ out) {
  int wid = (blockIdx.x * blockDim.x + threadIdx.x) >> 5;
  int lane = threadIdx.x & 31;
  if (wid >= NBATCH * 512) return;
  int b = wid >> 9, tt = wid & 511;
  const float* x = h + ((size_t)b * LTOT + 1536 + tt) * DMODEL;
  float s = 0.f;
  #pragma unroll
  for (int j = 0; j < 16; ++j) s += x[lane + j * 32] * pw[lane + j * 32];
  for (int o = 16; o >= 1; o >>= 1) s += __shfl_xor(s, o, 32);
  if (lane == 0) out[wid] = s + pb[0];
}

// ---------- host ----------
extern "C" void kernel_launch(void* const* d_in, const int* in_sizes, int n_in,
                              void* d_out, int out_size, void* d_ws, size_t ws_size,
                              hipStream_t stream) {
  const float* x_enc   = (const float*)d_in[0];
  const float* xm_enc  = (const float*)d_in[1];
  const float* x_dec   = (const float*)d_in[2];
  const float* xm_dec  = (const float*)d_in[3];
  const float* token_w = (const float*)d_in[4];
  const float* temp_w  = (const float*)d_in[5];
  const float* qk_w    = (const float*)d_in[6];
  const float* v_w     = (const float*)d_in[7];
  const float* out_w   = (const float*)d_in[8];
  const float* out_b   = (const float*)d_in[9];
  const float* ln1_g   = (const float*)d_in[10];
  const float* ln1_b   = (const float*)d_in[11];
  const float* ff1_w   = (const float*)d_in[12];
  const float* ff1_b   = (const float*)d_in[13];
  const float* ff2_w   = (const float*)d_in[14];
  const float* ff2_b   = (const float*)d_in[15];
  const float* ln2_g   = (const float*)d_in[16];
  const float* ln2_b   = (const float*)d_in[17];
  const float* norm_g  = (const float*)d_in[18];
  const float* norm_b  = (const float*)d_in[19];
  const float* proj_w  = (const float*)d_in[20];
  const float* proj_b  = (const float*)d_in[21];
  const float* rotations = (const float*)d_in[22];

  char* ws = (char*)d_ws;
  unsigned short* wbf = (unsigned short*)ws;                  // bf16 weights (5767168 elems)
  float* h    = (float*)(ws + 11534336);
  float* act1 = (float*)(ws + 45088768);                      // qk activations fp32
  float* act2 = (float*)(ws + 78643200);                      // v activations fp32
  float* act3 = (float*)(ws + 112197632);                     // ffn pre-LN fp32
  float* big  = (float*)(ws + 145752064);                     // aliased: so fp32 OR y bf16
  float* slog = (float*)(ws + 279969792);
  int* buckets = (int*)(ws + 282066944);
  int* st      = (int*)(ws + 284164096);
  int* undo    = (int*)(ws + 286261248);
  int* startb  = (int*)(ws + 288358400);
  unsigned short* hbf    = (unsigned short*)(ws + 288882688); // bf16 copy of h
  unsigned short* attnbf = (unsigned short*)(ws + 305659904); // bf16 merged attn
  unsigned short* ybf    = (unsigned short*)big;              // bf16 FFN intermediate (aliased)
  (void)in_sizes; (void)n_in; (void)out_size; (void)ws_size;

  cvt_bf16_kernel<<<(524288 + 255) / 256, 256, 0, stream>>>(qk_w,  wbf + 0,       524288);
  cvt_bf16_kernel<<<(524288 + 255) / 256, 256, 0, stream>>>(v_w,   wbf + 524288,  524288);
  cvt_bf16_kernel<<<(524288 + 255) / 256, 256, 0, stream>>>(out_w, wbf + 1048576, 524288);
  cvt_bf16_kernel<<<(2097152 + 255) / 256, 256, 0, stream>>>(ff1_w, wbf + 1572864, 2097152);
  cvt_bf16_kernel<<<(2097152 + 255) / 256, 256, 0, stream>>>(ff2_w, wbf + 3670016, 2097152);

  embed_kernel<<<MROWS, 128, 0, stream>>>(x_enc, x_dec, xm_enc, xm_dec, token_w, temp_w, h, hbf);

  for (int l = 0; l < 2; ++l) {
    const unsigned short* wqk  = wbf + 0       + l * 262144;
    const unsigned short* wv   = wbf + 524288  + l * 262144;
    const unsigned short* wout = wbf + 1048576 + l * 262144;
    const unsigned short* wff1 = wbf + 1572864 + l * 1048576;
    const unsigned short* wff2 = wbf + 3670016 + l * 1048576;

    gemm_bf16_tdm<<<dim3(DMODEL / 64, MROWS / 128), 256, 0, stream>>>(
        hbf, wqk, act1, nullptr, nullptr, nullptr, MROWS, DMODEL, DMODEL, 0);
    gemm_bf16_tdm<<<dim3(DMODEL / 64, MROWS / 128), 256, 0, stream>>>(
        hbf, wv, act2, nullptr, nullptr, nullptr, MROWS, DMODEL, DMODEL, 0);

    hash_kernel<<<NBH * LTOT, 128, 0, stream>>>(act1, rotations + (size_t)l * 65536, buckets);
    hist_kernel<<<NBH, 256, 0, stream>>>(buckets, startb);
    scatter_kernel<<<(NBH * 2048) / 256, 256, 0, stream>>>(buckets, startb, st, undo);

    chunk_attn_kernel<<<(NBH * NCHUNK) / 8, 256, 0, stream>>>(act1, act2, st, big, slog);
    combine_kernel<<<(NBH * LTOT) / 256, 256, 0, stream>>>(big, slog, undo, attnbf);

    gemm_bf16_tdm<<<dim3(DMODEL / 64, MROWS / 128), 256, 0, stream>>>(
        attnbf, wout, h, nullptr, out_b + l * DMODEL, h, MROWS, DMODEL, DMODEL, 3);
    ln_kernel<<<(MROWS * 32) / 256, 256, 0, stream>>>(h, h, hbf, ln1_g + l * DMODEL, ln1_b + l * DMODEL, MROWS);

    gemm_bf16_tdm<<<dim3(DFF / 64, MROWS / 128), 256, 0, stream>>>(
        hbf, wff1, nullptr, ybf, ff1_b + l * DFF, nullptr, MROWS, DFF, DMODEL, 13);
    gemm_bf16_tdm<<<dim3(DMODEL / 64, MROWS / 128), 256, 0, stream>>>(
        ybf, wff2, act3, nullptr, ff2_b + l * DMODEL, h, MROWS, DMODEL, DFF, 3);
    ln_kernel<<<(MROWS * 32) / 256, 256, 0, stream>>>(act3, h, hbf, ln2_g + l * DMODEL, ln2_b + l * DMODEL, MROWS);
  }

  ln_kernel<<<(MROWS * 32) / 256, 256, 0, stream>>>(h, h, nullptr, norm_g, norm_b, MROWS);
  proj_kernel<<<(NBATCH * 512 * 32) / 256, 256, 0, stream>>>(h, proj_w, proj_b, (float*)d_out);
}
